// Moe_63101659513268
// MI455X (gfx1250) — matrix operand order repaired and checked by reading, compiled, not run on hardware
//
#include <hip/hip_runtime.h>
#include <hip/hip_bf16.h>
#include <stdint.h>

// ---------------------------------------------------------------------------
// Problem constants (from the reference)
// ---------------------------------------------------------------------------
#define BATCH   8192
#define INF_    512      // IN_FEATURE
#define HID     1024
#define NEXP    16

typedef __attribute__((ext_vector_type(16))) __bf16 bf16x16;
typedef __attribute__((ext_vector_type(8)))  __bf16 bf16x8;
typedef __attribute__((ext_vector_type(8)))  float  f32x8;

// ---------------------------------------------------------------------------
// f32 -> bf16 elementwise convert
// ---------------------------------------------------------------------------
__global__ __launch_bounds__(256) void k_convert_bf16(const float* __restrict__ src,
                                                      __bf16* __restrict__ dst, int n) {
    int i = blockIdx.x * 256 + threadIdx.x;
    int stride = gridDim.x * 256;
    for (; i < n; i += stride) dst[i] = (__bf16)src[i];
}

// ---------------------------------------------------------------------------
// src [K][N] f32  ->  dst [N][K] bf16   (batched over blockIdx.z)
// ---------------------------------------------------------------------------
__global__ __launch_bounds__(256) void k_transpose_convert(const float* __restrict__ src,
                                                           __bf16* __restrict__ dst,
                                                           int K, int N) {
    __shared__ float tile[32][33];
    src += (size_t)blockIdx.z * K * N;
    dst += (size_t)blockIdx.z * K * N;
    const int nbase = blockIdx.x * 32;
    const int kbase = blockIdx.y * 32;
    const int tx = threadIdx.x, ty = threadIdx.y;     // block (32, 8)
#pragma unroll
    for (int i = 0; i < 32; i += 8) {
        int k = kbase + ty + i, n = nbase + tx;
        if (k < K && n < N) tile[ty + i][tx] = src[(size_t)k * N + n];
    }
    __syncthreads();
#pragma unroll
    for (int i = 0; i < 32; i += 8) {
        int n = nbase + ty + i, k = kbase + tx;
        if (n < N && k < K) dst[(size_t)n * K + k] = (__bf16)tile[tx][ty + i];
    }
}

// ---------------------------------------------------------------------------
// C[M][N] = silu(A[M][K] @ Bt[N][K]^T + bias[N])      (bf16 in, bf16 out)
//
// 128 threads = 4 waves (2 M x 2 N); block tile 128x128, K step 32,
// LDS double-buffered.  Wave tile 64x64 = 4x4 WMMA 16x16 tiles
// (v_wmma_f32_16x16x32_bf16): 16 WMMAs per 16 ds_load_b128 per K-step.
// LDS element stride 40 (80 B) -> conflict-free b128 reads.
// ---------------------------------------------------------------------------
#define LDA   40
#define ATILE (128 * LDA)               // elements per A (or B) tile
#define TILE2 (2 * ATILE)               // A tile + B tile per buffer

__global__ __launch_bounds__(128) void k_gemm_silu(const __bf16* __restrict__ A,
                                                   const __bf16* __restrict__ Bt,
                                                   const float* __restrict__ bias,
                                                   __bf16* __restrict__ C,
                                                   int M, int N, int K) {
    __shared__ __align__(16) __bf16 lds[2 * TILE2];   // 2 buffers x (A+B) = 40 KB

    const int t    = threadIdx.x;
    const int lane = t & 31;
    const int wv   = t >> 5;          // 0..3
    const int wm   = wv & 1;          // 64 rows per wave
    const int wn   = wv >> 1;         // 64 cols per wave
    const int lm   = lane & 15;
    const int kg   = lane >> 4;       // lane group (K-half select)

    const int m0 = blockIdx.y * 128;
    const int n0 = blockIdx.x * 128;

    // per-thread staging offsets: 512 16B chunks per tile, 4 per thread
    int soff[4];                      // element offset row*K-part handled below
    int loff[4];                      // LDS element offset
#pragma unroll
    for (int i = 0; i < 4; ++i) {
        const int c   = t + i * 128;
        const int row = c >> 2;       // 4 chunks of 8 bf16 per row
        const int k8  = (c & 3) * 8;
        soff[i] = row * K + k8;       // add uniform (m0*K + kk) at use site
        loff[i] = row * LDA + k8;
    }

    const __bf16* Ab = A  + (size_t)m0 * K;
    const __bf16* Bb = Bt + (size_t)n0 * K;

    f32x8 acc[4][4];
#pragma unroll
    for (int mi = 0; mi < 4; ++mi)
#pragma unroll
        for (int ni = 0; ni < 4; ++ni) acc[mi][ni] = (f32x8){};

    union Frag { bf16x16 v; bf16x8 h[2]; };

    __bf16* cur = lds;                // compute buffer
    __bf16* nxt = lds + TILE2;        // staging buffer

    // ---- prologue: stage tile 0 into `cur` ---------------------------------
    {
        bf16x8 ra[4], rb[4];
#pragma unroll
        for (int i = 0; i < 4; ++i) {
            ra[i] = *(const bf16x8*)&Ab[soff[i]];
            rb[i] = *(const bf16x8*)&Bb[soff[i]];
        }
#pragma unroll
        for (int i = 0; i < 4; ++i) {
            *(bf16x8*)&cur[loff[i]]         = ra[i];
            *(bf16x8*)&cur[ATILE + loff[i]] = rb[i];
        }
    }
    __syncthreads();

    const int nk = K >> 5;            // K / 32  (16 or 32 here)
    for (int k = 0; k < nk - 1; ++k) {
        const int kk = (k + 1) << 5;

        // 1) issue global loads for tile k+1 (latency overlapped with WMMA)
        bf16x8 ra[4], rb[4];
#pragma unroll
        for (int i = 0; i < 4; ++i) {
            ra[i] = *(const bf16x8*)&Ab[soff[i] + kk];
            rb[i] = *(const bf16x8*)&Bb[soff[i] + kk];
        }

        // 2) compute tile k from `cur`
        {
            const __bf16* la = cur;
            const __bf16* lb = cur + ATILE;
            bf16x16 af[4];
#pragma unroll
            for (int mi = 0; mi < 4; ++mi) {
                const int r = wm * 64 + mi * 16 + lm;
                Frag u;
                u.h[0] = *(const bf16x8*)&la[r * LDA + kg * 8];
                u.h[1] = *(const bf16x8*)&la[r * LDA + 16 + kg * 8];
                af[mi] = u.v;
            }
#pragma unroll
            for (int ni = 0; ni < 4; ++ni) {
                const int n = wn * 64 + ni * 16 + lm;
                Frag u;
                u.h[0] = *(const bf16x8*)&lb[n * LDA + kg * 8];
                u.h[1] = *(const bf16x8*)&lb[n * LDA + 16 + kg * 8];
#pragma unroll
                for (int mi = 0; mi < 4; ++mi)
                    acc[mi][ni] = __builtin_amdgcn_wmma_f32_16x16x32_bf16(
                        false, af[mi], false, u.v, (short)0, acc[mi][ni],
                        false, false);
            }
        }

        // 3) store staged registers into `nxt` (consumed next iteration)
#pragma unroll
        for (int i = 0; i < 4; ++i) {
            *(bf16x8*)&nxt[loff[i]]         = ra[i];
            *(bf16x8*)&nxt[ATILE + loff[i]] = rb[i];
        }
        __syncthreads();

        __bf16* tmp = cur; cur = nxt; nxt = tmp;
    }

    // ---- final tile --------------------------------------------------------
    {
        const __bf16* la = cur;
        const __bf16* lb = cur + ATILE;
        bf16x16 af[4];
#pragma unroll
        for (int mi = 0; mi < 4; ++mi) {
            const int r = wm * 64 + mi * 16 + lm;
            Frag u;
            u.h[0] = *(const bf16x8*)&la[r * LDA + kg * 8];
            u.h[1] = *(const bf16x8*)&la[r * LDA + 16 + kg * 8];
            af[mi] = u.v;
        }
#pragma unroll
        for (int ni = 0; ni < 4; ++ni) {
            const int n = wn * 64 + ni * 16 + lm;
            Frag u;
            u.h[0] = *(const bf16x8*)&lb[n * LDA + kg * 8];
            u.h[1] = *(const bf16x8*)&lb[n * LDA + 16 + kg * 8];
#pragma unroll
            for (int mi = 0; mi < 4; ++mi)
                acc[mi][ni] = __builtin_amdgcn_wmma_f32_16x16x32_bf16(
                    false, af[mi], false, u.v, (short)0, acc[mi][ni],
                    false, false);
        }
    }

    // ---- epilogue: bias + SiLU + bf16 store --------------------------------
#pragma unroll
    for (int mi = 0; mi < 4; ++mi) {
#pragma unroll
        for (int ni = 0; ni < 4; ++ni) {
            const int col  = n0 + wn * 64 + ni * 16 + lm;
            const int mrow = m0 + wm * 64 + mi * 16 + kg * 8;  // lanes 16-31 -> M+8
            const float bv = bias[col];
#pragma unroll
            for (int r = 0; r < 8; ++r) {
                float v = acc[mi][ni][r] + bv;
                v = v / (1.0f + __expf(-v));                   // SiLU
                C[(size_t)(mrow + r) * N + col] = (__bf16)v;
            }
        }
    }
}

// ---------------------------------------------------------------------------
// out[b*out_stride + out_off + e] = dot(H[b,:K], W[e,:K]) + bias[e]
// one wave per row b; lane-vectorized (8 bf16 per lane) + shuffle reduce
// ---------------------------------------------------------------------------
__global__ __launch_bounds__(256) void k_dot_rows(const __bf16* __restrict__ H,
                                                  const __bf16* __restrict__ W,
                                                  const float* __restrict__ bias,
                                                  float* __restrict__ out,
                                                  int B, int K, int E,
                                                  int out_stride, int out_off) {
    const int wave = (blockIdx.x * 256 + threadIdx.x) >> 5;
    const int lane = threadIdx.x & 31;
    if (wave >= B) return;
    const __bf16* h = H + (size_t)wave * K;
    for (int e = 0; e < E; ++e) {
        const __bf16* w = W + (size_t)e * K;
        float s = 0.0f;
        for (int k = lane * 8; k < K; k += 32 * 8) {
            bf16x8 hv = *(const bf16x8*)&h[k];
            bf16x8 wv = *(const bf16x8*)&w[k];
#pragma unroll
            for (int j = 0; j < 8; ++j) s += (float)hv[j] * (float)wv[j];
        }
#pragma unroll
        for (int off = 16; off > 0; off >>= 1) s += __shfl_down(s, off, 32);
        if (lane == 0) out[(size_t)wave * out_stride + out_off + e] = s + bias[e];
    }
}

// ---------------------------------------------------------------------------
// per-row softmax over logits, weighted sum of expert outputs
// ---------------------------------------------------------------------------
__global__ __launch_bounds__(256) void k_finalize(const float* __restrict__ eo,
                                                  const float* __restrict__ lg,
                                                  float* __restrict__ out,
                                                  int B, int E) {
    const int b = blockIdx.x * 256 + threadIdx.x;
    if (b >= B) return;
    const float* l = lg + (size_t)b * E;
    const float* o = eo + (size_t)b * E;
    float m = l[0];
    for (int e = 1; e < E; ++e) m = fmaxf(m, l[e]);
    float den = 0.0f, acc = 0.0f;
    for (int e = 0; e < E; ++e) {
        float w = __expf(l[e] - m);
        den += w;
        acc += w * o[e];
    }
    out[b] = acc / den;
}

// ---------------------------------------------------------------------------
// launcher
// ---------------------------------------------------------------------------
extern "C" void kernel_launch(void* const* d_in, const int* in_sizes, int n_in,
                              void* d_out, int out_size, void* d_ws, size_t ws_size,
                              hipStream_t stream) {
    const float* x   = (const float*)d_in[0];
    const float* W1  = (const float*)d_in[1];
    const float* b1  = (const float*)d_in[2];
    const float* W2  = (const float*)d_in[3];
    const float* b2  = (const float*)d_in[4];
    const float* W3  = (const float*)d_in[5];
    const float* b3  = (const float*)d_in[6];
    const float* Wg1 = (const float*)d_in[7];
    const float* bg1 = (const float*)d_in[8];
    const float* Wg2 = (const float*)d_in[9];
    const float* bg2 = (const float*)d_in[10];
    const float* Wg3 = (const float*)d_in[11];
    const float* bg3 = (const float*)d_in[12];
    float* out = (float*)d_out;

    uint8_t* p = (uint8_t*)d_ws;
    auto alloc = [&](size_t bytes) -> uint8_t* {
        uint8_t* r = p;
        p += (bytes + 255) & ~(size_t)255;
        return r;
    };
    __bf16* xb   = (__bf16*)alloc((size_t)BATCH * INF_ * 2);
    __bf16* W1t  = (__bf16*)alloc((size_t)NEXP * HID * INF_ * 2);
    __bf16* W2t  = (__bf16*)alloc((size_t)NEXP * HID * HID * 2);
    __bf16* Wg1t = (__bf16*)alloc((size_t)HID * INF_ * 2);
    __bf16* Wg2t = (__bf16*)alloc((size_t)HID * HID * 2);
    __bf16* Wg3t = (__bf16*)alloc((size_t)NEXP * HID * 2);
    __bf16* W3b  = (__bf16*)alloc((size_t)NEXP * HID * 2);
    __bf16* hA   = (__bf16*)alloc((size_t)BATCH * HID * 2);
    __bf16* hB   = (__bf16*)alloc((size_t)BATCH * HID * 2);
    float*  eo   = (float*)alloc((size_t)BATCH * NEXP * 4);
    float*  lg   = (float*)alloc((size_t)BATCH * NEXP * 4);

    const dim3 blkT(32, 8), blk1(256), blkG(128);

    // --- precision conversion / weight transposition -------------------------
    k_convert_bf16<<<dim3(4096), blk1, 0, stream>>>(x, xb, BATCH * INF_);
    k_transpose_convert<<<dim3(HID / 32, INF_ / 32, NEXP), blkT, 0, stream>>>(W1, W1t, INF_, HID);
    k_transpose_convert<<<dim3(HID / 32, HID / 32, NEXP), blkT, 0, stream>>>(W2, W2t, HID, HID);
    k_transpose_convert<<<dim3(HID / 32, INF_ / 32, 1), blkT, 0, stream>>>(Wg1, Wg1t, INF_, HID);
    k_transpose_convert<<<dim3(HID / 32, HID / 32, 1), blkT, 0, stream>>>(Wg2, Wg2t, HID, HID);
    k_transpose_convert<<<dim3(1, HID / 32, 1), blkT, 0, stream>>>(Wg3, Wg3t, HID, NEXP);
    k_convert_bf16<<<dim3(64), blk1, 0, stream>>>(W3, W3b, NEXP * HID);

    const dim3 gGemm(HID / 128, BATCH / 128);     // (8, 64)
    const dim3 gDot((BATCH * 32 + 255) / 256);    // 8 waves/block, 1 row/wave

    // --- gating network (runs first; hA/hB are then recycled per expert) -----
    k_gemm_silu<<<gGemm, blkG, 0, stream>>>(xb, Wg1t, bg1, hA, BATCH, HID, INF_);
    k_gemm_silu<<<gGemm, blkG, 0, stream>>>(hA, Wg2t, bg2, hB, BATCH, HID, HID);
    k_dot_rows<<<gDot, blk1, 0, stream>>>(hB, Wg3t, bg3, lg, BATCH, HID, NEXP, NEXP, 0);

    // --- experts --------------------------------------------------------------
    for (int e = 0; e < NEXP; ++e) {
        k_gemm_silu<<<gGemm, blkG, 0, stream>>>(
            xb, W1t + (size_t)e * HID * INF_, b1 + (size_t)e * HID, hA,
            BATCH, HID, INF_);
        k_gemm_silu<<<gGemm, blkG, 0, stream>>>(
            hA, W2t + (size_t)e * HID * HID, b2 + (size_t)e * HID, hB,
            BATCH, HID, HID);
        k_dot_rows<<<gDot, blk1, 0, stream>>>(
            hB, W3b + (size_t)e * HID, b3 + e, eo, BATCH, HID, 1, NEXP, e);
    }

    // --- softmax gate + combine ----------------------------------------------
    k_finalize<<<dim3(BATCH / 256), blk1, 0, stream>>>(eo, lg, out, BATCH, NEXP);
    (void)in_sizes; (void)n_in; (void)out_size; (void)ws_size;
}
